// LSTM_71167608094989
// MI455X (gfx1250) — matrix operand order repaired and checked by reading, not compiled, not run on hardware
//
#include <hip/hip_runtime.h>
#include <hip/hip_bf16.h>

// Problem constants (reference: T, B, I, H = 512, 64, 1024, 1024)
#define T_STEPS 512
#define BATCH   64
#define IN_DIM  1024
#define H_DIM   1024

typedef __attribute__((ext_vector_type(16))) __bf16 bf16x16;
typedef __attribute__((ext_vector_type(8)))  __bf16 bf16x8;
typedef __attribute__((ext_vector_type(8)))  float  floatx8;

// ---------------------------------------------------------------------------
// Prep kernels
// ---------------------------------------------------------------------------
__global__ void f32_to_bf16_kernel(const float* __restrict__ src,
                                   __bf16* __restrict__ dst, int n) {
    int i = blockIdx.x * blockDim.x + threadIdx.x;
    int s = gridDim.x * blockDim.x;
    for (; i < n; i += s) dst[i] = (__bf16)src[i];
}

__global__ void bias_combine_kernel(const float* __restrict__ a,
                                    const float* __restrict__ b,
                                    float* __restrict__ out, int n) {
    int i = blockIdx.x * blockDim.x + threadIdx.x;
    if (i < n) out[i] = a[i] + b[i];
}

__global__ void zero_words_kernel(unsigned int* __restrict__ p, int nwords) {
    int i = blockIdx.x * blockDim.x + threadIdx.x;
    int s = gridDim.x * blockDim.x;
    for (; i < nwords; i += s) p[i] = 0u;
}

// ---------------------------------------------------------------------------
// WMMA fragment loaders (layouts per CDNA5 ISA 7.12.2, bf16 16x16x32)
//   A 16x32: lane<16 holds row=lane, K = {kc..kc+7} U {kc+16..kc+23}
//            lane>=16 holds row=lane-16, K shifted by +8 in each half
//   B 32x16: lane<16 holds col=lane, K = {kc..kc+7} U {kc+16..kc+23}
//            lane>=16 holds col=lane-16, K shifted by +8 in each half
// ---------------------------------------------------------------------------
__device__ __forceinline__ bf16x16 load_a_frag(const __bf16* __restrict__ row,
                                               int kc, int half) {
    bf16x8 lo = *reinterpret_cast<const bf16x8*>(row + kc + half * 8);
    bf16x8 hi = *reinterpret_cast<const bf16x8*>(row + kc + 16 + half * 8);
    bf16x16 r;
#pragma unroll
    for (int i = 0; i < 8; ++i) { r[i] = lo[i]; r[i + 8] = hi[i]; }
    return r;
}

__device__ __forceinline__ bf16x16 load_b_frag(const __bf16* __restrict__ wrow,
                                               int kc, int half) {
    bf16x8 lo = *reinterpret_cast<const bf16x8*>(wrow + kc + half * 8);
    bf16x8 hi = *reinterpret_cast<const bf16x8*>(wrow + kc + 16 + half * 8);
    bf16x16 r;
#pragma unroll
    for (int i = 0; i < 8; ++i) { r[i] = lo[i]; r[i + 8] = hi[i]; }
    return r;
}

// ---------------------------------------------------------------------------
// One LSTM timestep, fully fused: gates GEMM (input + recurrent, bf16 WMMA,
// fp32 accum) -> LDS -> cell elementwise update.
// grid = (4 batch-tiles, 64 n-tiles), block = 128 threads = 4 waves,
// wave w computes gate w (i,f,g,o) for its block's 16x16 tile.
// ---------------------------------------------------------------------------
__global__ __launch_bounds__(128)
void lstm_step_kernel(const __bf16* __restrict__ xt,     // [BATCH, IN_DIM]
                      const __bf16* __restrict__ w_ih,   // [4H, IN_DIM]
                      const __bf16* __restrict__ w_hh,   // [4H, H]
                      const float*  __restrict__ bias,   // [4H] (b_ih + b_hh)
                      const __bf16* __restrict__ h_prev, // [BATCH, H]
                      float*        __restrict__ c_state,// [BATCH, H] in-place
                      __bf16*       __restrict__ h_next, // [BATCH, H]
                      float*        __restrict__ out_f32)// [BATCH, H] or null
{
    const int wave  = threadIdx.x >> 5;   // gate index 0..3
    const int lane  = threadIdx.x & 31;
    const int half  = lane >> 4;
    const int l15   = lane & 15;
    const int btile = blockIdx.x;         // 0..3  (16 batch rows each)
    const int ntile = blockIdx.y;         // 0..63 (16 hidden cols each)

    const int arow = btile * 16 + l15;                    // batch row
    const int grow = wave * H_DIM + ntile * 16 + l15;     // W row == gate col

    const __bf16* a_x  = xt     + (size_t)arow * IN_DIM;
    const __bf16* a_h  = h_prev + (size_t)arow * H_DIM;
    const __bf16* b_ih = w_ih   + (size_t)grow * IN_DIM;
    const __bf16* b_hh = w_hh   + (size_t)grow * H_DIM;

    floatx8 acc = {};

    // input projection contribution: K = IN_DIM
#pragma unroll 4
    for (int kc = 0; kc < IN_DIM; kc += 32) {
        bf16x16 a = load_a_frag(a_x, kc, half);
        bf16x16 b = load_b_frag(b_ih, kc, half);
        acc = __builtin_amdgcn_wmma_f32_16x16x32_bf16(
            false, a, false, b, (short)0, acc, false, false);
    }
    // recurrent contribution: K = H_DIM
#pragma unroll 4
    for (int kc = 0; kc < H_DIM; kc += 32) {
        bf16x16 a = load_a_frag(a_h, kc, half);
        bf16x16 b = load_b_frag(b_hh, kc, half);
        acc = __builtin_amdgcn_wmma_f32_16x16x32_bf16(
            false, a, false, b, (short)0, acc, false, false);
    }

    // C/D layout: VGPR r, lanes 0-15 -> row r, lanes 16-31 -> row r+8
    __shared__ float lds_g[4][16][16];
#pragma unroll
    for (int r = 0; r < 8; ++r)
        lds_g[wave][r + 8 * half][l15] = acc[r];
    __syncthreads();

    // Cell update: 256 tile elements, 128 threads -> 2 each
    for (int e = threadIdx.x; e < 256; e += 128) {
        const int row   = e >> 4;
        const int col   = e & 15;
        const int b_idx = btile * 16 + row;
        const int n     = ntile * 16 + col;

        float gi = lds_g[0][row][col] + bias[0 * H_DIM + n];
        float gf = lds_g[1][row][col] + bias[1 * H_DIM + n];
        float gg = lds_g[2][row][col] + bias[2 * H_DIM + n];
        float go = lds_g[3][row][col] + bias[3 * H_DIM + n];

        float si = 1.0f / (1.0f + __expf(-gi));
        float sf = 1.0f / (1.0f + __expf(-gf));
        float so = 1.0f / (1.0f + __expf(-go));
        float tg = tanhf(gg);

        const size_t idx = (size_t)b_idx * H_DIM + n;
        float cn = sf * c_state[idx] + si * tg;   // tile-private: in-place OK
        c_state[idx] = cn;
        float hn = so * tanhf(cn);
        h_next[idx] = (__bf16)hn;
        if (out_f32) out_f32[idx] = hn;
    }
}

// ---------------------------------------------------------------------------
// Host-side orchestration
// ---------------------------------------------------------------------------
extern "C" void kernel_launch(void* const* d_in, const int* in_sizes, int n_in,
                              void* d_out, int out_size, void* d_ws, size_t ws_size,
                              hipStream_t stream) {
    (void)in_sizes; (void)n_in; (void)out_size; (void)ws_size;

    const float* input_ = (const float*)d_in[0];
    const float* w_ih0  = (const float*)d_in[1];
    const float* w_hh0  = (const float*)d_in[2];
    const float* b_ih0  = (const float*)d_in[3];
    const float* b_hh0  = (const float*)d_in[4];
    const float* w_ih1  = (const float*)d_in[5];
    const float* w_hh1  = (const float*)d_in[6];
    const float* b_ih1  = (const float*)d_in[7];
    const float* b_hh1  = (const float*)d_in[8];
    float* out = (float*)d_out;

    // Workspace carve-out (256B aligned chunks), ~161 MB total
    char* ws = (char*)d_ws;
    size_t off = 0;
    auto carve = [&](size_t bytes) -> void* {
        off = (off + 255) & ~(size_t)255;
        void* p = ws + off;
        off += bytes;
        return p;
    };

    const size_t NX = (size_t)T_STEPS * BATCH * IN_DIM;   // 33.5M
    const size_t NW = (size_t)4 * H_DIM * IN_DIM;         // 4.2M per matrix
    const size_t NS = (size_t)BATCH * H_DIM;              // 65536 per state

    __bf16* x_bf    = (__bf16*)carve(NX * 2);
    __bf16* wih0_bf = (__bf16*)carve(NW * 2);
    __bf16* whh0_bf = (__bf16*)carve(NW * 2);
    __bf16* wih1_bf = (__bf16*)carve(NW * 2);
    __bf16* whh1_bf = (__bf16*)carve(NW * 2);
    __bf16* h1_bf   = (__bf16*)carve((size_t)T_STEPS * NS * 2);
    float*  bias0   = (float*) carve((size_t)4 * H_DIM * 4);
    float*  bias1   = (float*) carve((size_t)4 * H_DIM * 4);
    __bf16* h_st0   = (__bf16*)carve(NS * 2);
    __bf16* h_st1   = (__bf16*)carve(NS * 2);
    float*  c_st    = (float*) carve(NS * 4);
    __bf16* h_zero  = (__bf16*)carve(NS * 2);

    // fp32 -> bf16 conversions
    f32_to_bf16_kernel<<<2048, 256, 0, stream>>>(input_, x_bf, (int)NX);
    f32_to_bf16_kernel<<<1024, 256, 0, stream>>>(w_ih0, wih0_bf, (int)NW);
    f32_to_bf16_kernel<<<1024, 256, 0, stream>>>(w_hh0, whh0_bf, (int)NW);
    f32_to_bf16_kernel<<<1024, 256, 0, stream>>>(w_ih1, wih1_bf, (int)NW);
    f32_to_bf16_kernel<<<1024, 256, 0, stream>>>(w_hh1, whh1_bf, (int)NW);

    // fused biases
    bias_combine_kernel<<<(4 * H_DIM + 255) / 256, 256, 0, stream>>>(
        b_ih0, b_hh0, bias0, 4 * H_DIM);
    bias_combine_kernel<<<(4 * H_DIM + 255) / 256, 256, 0, stream>>>(
        b_ih1, b_hh1, bias1, 4 * H_DIM);

    const dim3 step_grid(4, 64);

    // ---------------- Layer 0 ----------------
    // zero h0 and c0
    zero_words_kernel<<<64, 256, 0, stream>>>((unsigned int*)h_zero, (int)(NS / 2));
    zero_words_kernel<<<64, 256, 0, stream>>>((unsigned int*)c_st,   (int)NS);

    for (int t = 0; t < T_STEPS; ++t) {
        const __bf16* hp = (t == 0) ? h_zero : (h1_bf + (size_t)(t - 1) * NS);
        __bf16* hn = h1_bf + (size_t)t * NS;   // h chains through h1 buffer
        lstm_step_kernel<<<step_grid, 128, 0, stream>>>(
            x_bf + (size_t)t * BATCH * IN_DIM,
            wih0_bf, whh0_bf, bias0,
            hp, c_st, hn, /*out_f32=*/nullptr);
    }

    // ---------------- Layer 1 ----------------
    zero_words_kernel<<<64, 256, 0, stream>>>((unsigned int*)h_st0, (int)(NS / 2));
    zero_words_kernel<<<64, 256, 0, stream>>>((unsigned int*)c_st,  (int)NS);

    for (int t = 0; t < T_STEPS; ++t) {
        const __bf16* hp = (t & 1) ? h_st1 : h_st0;
        __bf16*       hn = (t & 1) ? h_st0 : h_st1;
        lstm_step_kernel<<<step_grid, 128, 0, stream>>>(
            h1_bf + (size_t)t * NS,
            wih1_bf, whh1_bf, bias1,
            hp, c_st, hn, out + (size_t)t * NS);
    }
}
